// LSTM_12292196401862
// MI455X (gfx1250) — matrix operand order repaired and checked by reading, not compiled, not run on hardware
//
#include <hip/hip_runtime.h>
#include <stdint.h>

// LSTM: B=64, T=1024, D=256, H=512, fp32 I/O, bf16 WMMA internals.
#define B_   64
#define T_   1024
#define D_   256
#define H_   512
#define G4H  2048   // 4*H

typedef __attribute__((ext_vector_type(16))) __bf16 v16bf;
typedef __attribute__((ext_vector_type(8)))  float  v8f;

union Frag16 { v16bf v; uint4 q[2]; };

// float -> bf16 (round to nearest even)
__device__ __forceinline__ unsigned short f2bf(float f) {
  unsigned u = __float_as_uint(f);
  if ((u & 0x7fffffffu) > 0x7f800000u) return (unsigned short)((u >> 16) | 0x40); // quiet NaN
  unsigned r = u + 0x7fffu + ((u >> 16) & 1u);
  return (unsigned short)(r >> 16);
}

__device__ __forceinline__ float sigmoid_f(float x) { return 1.0f / (1.0f + __expf(-x)); }
__device__ __forceinline__ float tanh_f(float x)    { return 1.0f - 2.0f / (__expf(2.0f * x) + 1.0f); }

// ---------------- setup kernels ----------------

__global__ void k_init(unsigned short* hbuf, unsigned int* ctr, int n) {
  int i = blockIdx.x * blockDim.x + threadIdx.x;
  if (i == 0) *ctr = 0u;
  for (; i < n; i += gridDim.x * blockDim.x) hbuf[i] = 0;  // h_{-1} = 0 (both ping-pong bufs)
}

__global__ void k_cvt_x(const float* __restrict__ X, unsigned short* __restrict__ Xb, int n) {
  int i = blockIdx.x * blockDim.x + threadIdx.x;
  for (; i < n; i += gridDim.x * blockDim.x) Xb[i] = f2bf(X[i]);
}

// Pre-swizzle a [K x 2048] fp32 weight into bf16 WMMA B-fragment layout:
//   idx = ((nt*KB + kb)*32 + lane)*16 + e
//   column n = nt*16 + (lane&15), row K = kb*32 + (e>>3)*16 + (lane>>4)*8 + (e&7)
// so a wave's B fragment is one contiguous 32B load per lane (coalesced 1KB).
template <int KB>
__global__ void k_swz(const float* __restrict__ Wsrc, unsigned short* __restrict__ Wswz, int n) {
  int i = blockIdx.x * blockDim.x + threadIdx.x;
  for (; i < n; i += gridDim.x * blockDim.x) {
    int e    = i & 15;
    int lane = (i >> 4) & 31;
    int kb   = (i >> 9) & (KB - 1);
    int nt   = (i >> 9) >> (KB == 16 ? 4 : 3);
    int K    = kb * 32 + (e >> 3) * 16 + (lane >> 4) * 8 + (e & 7);
    int col  = nt * 16 + (lane & 15);
    Wswz[i]  = f2bf(Wsrc[(size_t)K * G4H + col]);
  }
}

// ---------------- phase 1: x_proj = X @ Wx + b, stored [T, B, 4H] ----------------
// 4096 M-tiles x 32 N-groups (4 N-tiles each) = 131072 waves = 16384 blocks.
__global__ __launch_bounds__(256) void k_xproj(const unsigned short* __restrict__ Xb,
                                               const unsigned short* __restrict__ WxS,
                                               const float* __restrict__ bias,
                                               float* __restrict__ xp) {
  const int lane = threadIdx.x & 31, wid = threadIdx.x >> 5;
  const int W   = blockIdx.x * 8 + wid;   // 0..131071
  const int mt  = W >> 5;                 // 0..4095
  const int ntg = W & 31;                 // 0..31 (covers 4 N-tiles)
  const int r   = lane & 15, kh = lane >> 4;
  const int rowA = mt * 16 + r;

  v8f acc[4];
  #pragma unroll
  for (int j = 0; j < 4; ++j)
    #pragma unroll
    for (int v = 0; v < 8; ++v) acc[j][v] = 0.0f;

  for (int kb = 0; kb < 8; ++kb) {        // K = 256 = 8 x 32
    Frag16 a;
    const uint4* ap = (const uint4*)(Xb + (size_t)rowA * D_ + kb * 32 + kh * 8);
    a.q[0] = ap[0];      // K = kb*32 + kh*8 + [0..7]
    a.q[1] = ap[2];      // K = kb*32 + 16 + kh*8 + [0..7]
    #pragma unroll
    for (int j = 0; j < 4; ++j) {
      const int nt = ntg * 4 + j;
      const uint4* bp = (const uint4*)WxS + ((size_t)(nt * 8 + kb) * 32 + lane) * 2;
      Frag16 b; b.q[0] = bp[0]; b.q[1] = bp[1];
      acc[j] = __builtin_amdgcn_wmma_f32_16x16x32_bf16(false, a.v, false, b.v,
                                                       (short)0, acc[j], false, false);
    }
  }
  #pragma unroll
  for (int j = 0; j < 4; ++j) {
    const int col = (ntg * 4 + j) * 16 + r;
    const float bv = bias[col];
    #pragma unroll
    for (int v = 0; v < 8; ++v) {
      const int R  = mt * 16 + kh * 8 + v;   // flat row of X = b*T + t
      const int bi = R >> 10, tt = R & 1023;
      xp[(size_t)(tt * B_ + bi) * G4H + col] = acc[j][v] + bv;
    }
  }
}

// ---------------- phase 2: persistent recurrence ----------------
// 32 blocks x 4 waves = 128 waves; block -> one H-column tile (ng), wave -> batch tile.
// Block's Wh slice (4 gates x 16 kb x 1KB = 64KB) cached in LDS ONCE via async copy;
// c stays in registers; h exchanged via bf16 ping-pong + device-wide atomic barrier;
// x_proj fragment for t+1 prefetched into registers during step t's WMMA loop.
__global__ __launch_bounds__(128) void k_lstm(const float* __restrict__ xp,
                                              const unsigned short* __restrict__ WhS,
                                              unsigned short* __restrict__ hbuf,
                                              float* __restrict__ out,
                                              unsigned int* __restrict__ ctr) {
  // 64 KB of B fragments, typed addrspace(3) array -> reads lower to ds_load_b128.
  __shared__ uint4 ldsW[4096];

  const int lane = threadIdx.x & 31, wid = threadIdx.x >> 5;
  const int ng = blockIdx.x;             // 0..31 -> H column tile
  const int m0 = wid * 16;               // batch tile
  const int n0 = ng * 16;
  const int r  = lane & 15, kh = lane >> 4;

  const int rowA  = m0 + r;              // h row for A fragments
  const int mBase = m0 + kh * 8;         // C/D row base
  const int colH  = n0 + r;              // column within H

  // ---- one-time async preload of this block's Wh slice into LDS ----
  // chunk idx = ((g*16+kb)*32 + lane2); 32B per chunk; 2048 chunks / 128 threads.
  // IMPORTANT: the LDS destination address is taken from &ldsW[...] (low 32 bits of
  // the flat address == byte offset in the workgroup's LDS allocation).  Passing the
  // array's address into the asm also makes ldsW escape, so the compiler cannot
  // treat it as never-written and fold the ds_load reads below to undef.
  for (int idx = threadIdx.x; idx < 2048; idx += 128) {
    const int g     = idx >> 9;
    const int kb    = (idx >> 5) & 15;
    const int lane2 = idx & 31;
    const unsigned short* src =
        WhS + (((size_t)(g * 32 + ng) * 16 + kb) * 32 + lane2) * 16;
    const unsigned lds0 = (unsigned)(uintptr_t)(&ldsW[2 * idx]);
    const unsigned lds1 = (unsigned)(uintptr_t)(&ldsW[2 * idx + 1]);
    asm volatile("global_load_async_to_lds_b128 %0, %1, off"
                 :: "v"(lds0), "v"(src) : "memory");
    asm volatile("global_load_async_to_lds_b128 %0, %1, off"
                 :: "v"(lds1), "v"(src + 8) : "memory");
  }
  asm volatile("s_wait_asynccnt 0x0" ::: "memory");
  __syncthreads();

  float c[8];
  #pragma unroll
  for (int v = 0; v < 8; ++v) c[v] = 0.0f;

  // x_proj register prefetch (t = 0)
  float xpR[4][8];
  #pragma unroll
  for (int g = 0; g < 4; ++g) {
    const size_t base = (size_t)mBase * G4H + g * H_ + colH;
    #pragma unroll
    for (int v = 0; v < 8; ++v) xpR[g][v] = xp[base + (size_t)v * G4H];
  }

  #pragma unroll 1
  for (int t = 0; t < T_; ++t) {
    const unsigned short* hR = hbuf + (size_t)(t & 1) * (B_ * H_);
    unsigned short*       hW = hbuf + (size_t)((t + 1) & 1) * (B_ * H_);

    // accumulators = prefetched x-projection (the +C of the GEMM)
    v8f acc[4];
    #pragma unroll
    for (int g = 0; g < 4; ++g)
      #pragma unroll
      for (int v = 0; v < 8; ++v) acc[g][v] = xpR[g][v];

    // prefetch x_proj for t+1 (hidden under the WMMA loop)
    if (t + 1 < T_) {
      #pragma unroll
      for (int g = 0; g < 4; ++g) {
        const size_t base = (size_t)((t + 1) * B_ + mBase) * G4H + g * H_ + colH;
        #pragma unroll
        for (int v = 0; v < 8; ++v) xpR[g][v] = xp[base + (size_t)v * G4H];
      }
    }

    // z += h_{t-1} @ Wh : K = 512 = 16 x 32; A pipelined 1 ahead; B from LDS.
    const uint4* apBase = (const uint4*)(hR + (size_t)rowA * H_);
    Frag16 aCur;
    aCur.q[0] = apBase[kh];          // kb=0: K = kh*8 + [0..7]
    aCur.q[1] = apBase[kh + 2];      //       K = 16 + kh*8 + [0..7]
    #pragma unroll 1
    for (int kb = 0; kb < 16; ++kb) {
      Frag16 aNext;
      const int nk = (kb + 1 < 16) ? kb + 1 : kb;
      aNext.q[0] = apBase[nk * 4 + kh];
      aNext.q[1] = apBase[nk * 4 + kh + 2];
      #pragma unroll
      for (int g = 0; g < 4; ++g) {
        const int fi = ((g * 16 + kb) * 32 + lane) * 2;
        Frag16 b;
        b.q[0] = ldsW[fi];       // ds_load_b128
        b.q[1] = ldsW[fi + 1];   // ds_load_b128
        acc[g] = __builtin_amdgcn_wmma_f32_16x16x32_bf16(false, aCur.v, false, b.v,
                                                         (short)0, acc[g], false, false);
      }
      aCur = aNext;
    }

    // gates + cell update (c lives in VGPRs for all 1024 steps)
    #pragma unroll
    for (int v = 0; v < 8; ++v) {
      const float ig = sigmoid_f(acc[0][v]);
      const float fg = sigmoid_f(acc[1][v]);
      const float gg = tanh_f(acc[2][v]);
      const float og = sigmoid_f(acc[3][v]);
      const float cn = fg * c[v] + ig * gg;
      c[v] = cn;
      const float hn = og * tanh_f(cn);
      const int m = mBase + v;
      out[((size_t)m * T_ + t) * H_ + colH] = hn;   // out[b][t][n]
      hW[(size_t)m * H_ + colH] = f2bf(hn);
    }

    // device-wide barrier: monotonic counter, release-add / acquire-spin
    __syncthreads();
    if (threadIdx.x == 0) {
      __hip_atomic_fetch_add(ctr, 1u, __ATOMIC_RELEASE, __HIP_MEMORY_SCOPE_AGENT);
      const unsigned target = (unsigned)gridDim.x * (unsigned)(t + 1);
      while (__hip_atomic_load(ctr, __ATOMIC_ACQUIRE, __HIP_MEMORY_SCOPE_AGENT) < target) {
        __builtin_amdgcn_s_sleep(1);
      }
    }
    __syncthreads();
  }
}

// ---------------- launch ----------------

extern "C" void kernel_launch(void* const* d_in, const int* in_sizes, int n_in,
                              void* d_out, int out_size, void* d_ws, size_t ws_size,
                              hipStream_t stream) {
  const float* X    = (const float*)d_in[0];  // [B,T,D]
  const float* Wx   = (const float*)d_in[1];  // [D,4H]
  const float* Wh   = (const float*)d_in[2];  // [H,4H]
  const float* bias = (const float*)d_in[3];  // [4H]
  float* out = (float*)d_out;                 // [B,T,H]

  char* ws = (char*)d_ws;
  size_t off = 0;
  float*          xpP  = (float*)(ws + off);          off += (size_t)T_ * B_ * G4H * 4; // 512 MB
  unsigned short* XbP  = (unsigned short*)(ws + off); off += (size_t)B_ * T_ * D_ * 2;  // 32 MB
  unsigned short* WxSP = (unsigned short*)(ws + off); off += (size_t)D_ * G4H * 2;      // 1 MB
  unsigned short* WhSP = (unsigned short*)(ws + off); off += (size_t)H_ * G4H * 2;      // 2 MB
  unsigned short* hbuf = (unsigned short*)(ws + off); off += (size_t)2 * B_ * H_ * 2;   // 128 KB
  unsigned int*   ctr  = (unsigned int*)(ws + off);   off += 256;

  k_init<<<64, 256, 0, stream>>>(hbuf, ctr, 2 * B_ * H_);
  k_cvt_x<<<2048, 256, 0, stream>>>(X, XbP, B_ * T_ * D_);
  k_swz<8><<<512, 256, 0, stream>>>(Wx, WxSP, D_ * G4H);
  k_swz<16><<<1024, 256, 0, stream>>>(Wh, WhSP, H_ * G4H);
  k_xproj<<<16384, 256, 0, stream>>>(XbP, WxSP, bias, xpP);
  k_lstm<<<32, 128, 0, stream>>>(xpP, WhSP, hbuf, out, ctr);
}
